// SurfEval_70660801954361
// MI455X (gfx1250) — compile-verified, matrix operand order checked by reading, not run on hardware
//
#include <hip/hip_runtime.h>
#include <hip/hip_bf16.h>

// ---------------------------------------------------------------------------
// NURBS surface eval, restructured as two batched dense GEMMs on the WMMA path.
//   B=16, M=N=64, P=Q=3, OUT=256x256, 3 output channels.
//   A[b]   : dense basis matrix [256 x 64] f16 (4 nonzeros/row) -- u and v share it
//   cpT    : [b][d][n][m] f16 (col-major B operand for stage 1)
//   T      : [b][d][u][n] f16 = A . cp_d
//   out_d  : T . A^T  -> out[b][u][v][d] f32
// ---------------------------------------------------------------------------

typedef __attribute__((ext_vector_type(16))) _Float16 v16h;
typedef __attribute__((ext_vector_type(8)))  _Float16 v8h;
typedef __attribute__((ext_vector_type(8)))  float    v8f;

#define NB   16
#define NCP  64
#define DEG  3
#define KLEN 68        // NCP + DEG + 1
#define OUTG 256
#define NCH  3

// ---------------------------------------------------------------------------
// Kernel 1: normalized knots, span search, Cox-de Boor, dense f16 basis rows.
// One block per batch, one thread per eval point t.
// ---------------------------------------------------------------------------
__global__ __launch_bounds__(OUTG) void basis_dense_kernel(
    const float* __restrict__ knot_u, _Float16* __restrict__ Au) {
  __shared__ float U[KLEN];
  const int b = blockIdx.x;

  if (threadIdx.x == 0) {
    const float* k = knot_u + b * KLEN;
    float arr[KLEN];
    float s = 0.f;
    for (int i = 0; i < KLEN; ++i) { s += k[i]; arr[i] = s; }
    const float lo = arr[0];
    const float inv = 1.0f / (arr[KLEN - 1] - lo);
    for (int i = 0; i < KLEN; ++i) U[i] = (arr[i] - lo) * inv;
  }
  __syncthreads();

  const int t = threadIdx.x;
  const float tv = 1e-5f + (1.0f - 2e-5f) * (float)t / (float)(OUTG - 1);

  // span = argmin over valid candidates (first-min, matching jnp.argmin)
  int best = 0;
  float bestv = 3.4e38f;
  for (int s = 0; s < KLEN - 2 * DEG; ++s) {
    const float dd = tv - U[DEG + s];
    const float cand = (dd > 1e-8f) ? dd : 1.0f;
    if (cand < bestv) { bestv = cand; best = s; }
  }
  const int span = best + DEG;

  // Cox-de Boor (deg 3), same fp ordering as the reference
  float Ni[DEG + 1];
  Ni[0] = 1.f; Ni[1] = 0.f; Ni[2] = 0.f; Ni[3] = 0.f;
  for (int kk = 1; kk <= DEG; ++kk) {
    float saved = 0.f;
    for (int r = 0; r < kk; ++r) {
      const float K1 = U[span + r + 1];
      const float K2 = U[span + 1 - kk + r];
      const float temp = Ni[r] / (K1 - tv + (tv - K2));
      Ni[r] = saved + (K1 - tv) * temp;
      saved = (tv - K2) * temp;
    }
    Ni[kk] = saved;
  }

  // scatter into a dense f16 row of length 64
  _Float16* row = Au + ((size_t)b * OUTG + t) * NCP;
  const int l0 = span - DEG;
  for (int j = 0; j < NCP; ++j) {
    const int l = j - l0;
    const float v = (l >= 0 && l <= DEG) ? Ni[l] : 0.f;
    row[j] = (_Float16)v;
  }
}

// ---------------------------------------------------------------------------
// Kernel 2: ctrl[b][m][n][4] f32 -> cpT[b][d][n][m] f16  (d = 0..2 only)
// ---------------------------------------------------------------------------
__global__ __launch_bounds__(256) void convert_cp_kernel(
    const float* __restrict__ ctrl, _Float16* __restrict__ cpT) {
  const int bd = blockIdx.x;
  const int b = bd / NCH, d = bd % NCH;
  const float* src = ctrl + (size_t)b * NCP * NCP * 4;
  _Float16* dst = cpT + (size_t)bd * NCP * NCP;
  for (int idx = threadIdx.x; idx < NCP * NCP; idx += blockDim.x) {
    const int n = idx >> 6, m = idx & 63;
    dst[n * NCP + m] = (_Float16)src[(m * NCP + n) * 4 + d];
  }
}

// ---------------------------------------------------------------------------
// 16x32 f16 tile load for WMMA operand (A row-major / B column-major, ld=K
// contiguous). Lane g=lane>>4, r=lane&15: halves[0..7]  = M[r][k0+8g .. +7],
// halves[8..15] = M[r][k0+16+8g .. +7]  (two 16-byte loads).
// ---------------------------------------------------------------------------
static __device__ __forceinline__ v16h load_tile16x32(
    const _Float16* __restrict__ base, int ld, int r0, int k0, int lane) {
  const int g = (lane >> 4) & 1;
  const int r = lane & 15;
  const _Float16* p = base + (size_t)(r0 + r) * ld + k0 + g * 8;
  const v8h lo = *(const v8h*)(p);
  const v8h hi = *(const v8h*)(p + 16);
  v16h out;
#pragma unroll
  for (int i = 0; i < 8; ++i) { out[i] = lo[i]; out[i + 8] = hi[i]; }
  return out;
}

// ---------------------------------------------------------------------------
// Kernel 3 (stage 1): T[b][d] (256x64) = Au[b] (256x64) . cp_d (64x64)
// One wave per 16x16 tile; K-loop = 2 x v_wmma_f32_16x16x32_f16.
// ---------------------------------------------------------------------------
__global__ __launch_bounds__(128) void stage1_kernel(
    const _Float16* __restrict__ Au, const _Float16* __restrict__ cpT,
    _Float16* __restrict__ T) {
  const int bd   = blockIdx.x >> 4;       // 48 (b,d) pairs, 16 blocks each
  const int tb   = blockIdx.x & 15;
  const int wave = threadIdx.x >> 5;
  const int lane = threadIdx.x & 31;
  const int tile = tb * 4 + wave;         // 0..63
  const int u0 = (tile >> 2) * 16;
  const int n0 = (tile & 3) * 16;
  const int b = bd / NCH;

  const _Float16* A  = Au  + (size_t)b  * OUTG * NCP;
  const _Float16* Bc = cpT + (size_t)bd * NCP * NCP;   // col-major (k contiguous)

  v8f acc = {};
#pragma unroll
  for (int k0 = 0; k0 < NCP; k0 += 32) {
    const v16h a  = load_tile16x32(A,  NCP, u0, k0, lane);
    const v16h bm = load_tile16x32(Bc, NCP, n0, k0, lane);
    acc = __builtin_amdgcn_wmma_f32_16x16x32_f16(
        false, a, false, bm, (short)0, acc, false, false);
  }

  _Float16* dst = T + (size_t)bd * OUTG * NCP;
  const int g = (lane >> 4) & 1, cc = lane & 15;
#pragma unroll
  for (int r = 0; r < 8; ++r)
    dst[(size_t)(u0 + g * 8 + r) * NCP + n0 + cc] = (_Float16)acc[r];
}

// ---------------------------------------------------------------------------
// Kernel 4 (stage 2): out_d (256x256) = T[b][d] (256x64) . Au[b]^T
// Au row-major [v][k] is exactly the column-major B operand (col v, k contig).
// Writes f32 output with channel-interleave stride 3.
// ---------------------------------------------------------------------------
__global__ __launch_bounds__(128) void stage2_kernel(
    const _Float16* __restrict__ T, const _Float16* __restrict__ Au,
    float* __restrict__ out) {
  const int bd   = blockIdx.x >> 6;       // 48 (b,d) pairs, 64 blocks each
  const int tb   = blockIdx.x & 63;
  const int wave = threadIdx.x >> 5;
  const int lane = threadIdx.x & 31;
  const int tile = tb * 4 + wave;         // 0..255
  const int u0 = (tile >> 4) * 16;
  const int v0 = (tile & 15) * 16;
  const int b = bd / NCH, d = bd % NCH;

  const _Float16* A  = T  + (size_t)bd * OUTG * NCP;
  const _Float16* Bc = Au + (size_t)b  * OUTG * NCP;

  v8f acc = {};
#pragma unroll
  for (int k0 = 0; k0 < NCP; k0 += 32) {
    const v16h a  = load_tile16x32(A,  NCP, u0, k0, lane);
    const v16h bm = load_tile16x32(Bc, NCP, v0, k0, lane);
    acc = __builtin_amdgcn_wmma_f32_16x16x32_f16(
        false, a, false, bm, (short)0, acc, false, false);
  }

  const int g = (lane >> 4) & 1, cc = lane & 15;
#pragma unroll
  for (int r = 0; r < 8; ++r) {
    const size_t u = (size_t)(u0 + g * 8 + r);
    out[(((size_t)b * OUTG + u) * OUTG + (v0 + cc)) * NCH + d] = acc[r];
  }
}

// ---------------------------------------------------------------------------
// Host launcher
// ---------------------------------------------------------------------------
extern "C" void kernel_launch(void* const* d_in, const int* in_sizes, int n_in,
                              void* d_out, int out_size, void* d_ws, size_t ws_size,
                              hipStream_t stream) {
  const float* ctrl   = (const float*)d_in[0];   // [16,64,64,4]
  const float* knot_u = (const float*)d_in[1];   // [16,68]
  // d_in[2] (knot_v) intentionally unused: reference builds V from knot_u.
  float* out = (float*)d_out;                    // [16,256,256,3]

  // Workspace carve-up (all 256B-aligned offsets):
  //   Au  f16 [16][256][64]     : 524288 B
  //   cpT f16 [16][3][64][64]   : 393216 B
  //   T   f16 [16][3][256][64]  : 1572864 B
  char* ws = (char*)d_ws;
  _Float16* Au  = (_Float16*)(ws);
  _Float16* cpT = (_Float16*)(ws + 524288);
  _Float16* T   = (_Float16*)(ws + 524288 + 393216);

  basis_dense_kernel<<<NB, OUTG, 0, stream>>>(knot_u, Au);
  convert_cp_kernel<<<NB * NCH, 256, 0, stream>>>(ctrl, cpT);
  stage1_kernel<<<NB * NCH * 16, 128, 0, stream>>>(Au, cpT, T);
  stage2_kernel<<<NB * NCH * 64, 128, 0, stream>>>(T, Au, out);
  (void)in_sizes; (void)n_in; (void)out_size; (void)ws_size;
}
